// FeedForwardQuantum_65481071399892
// MI455X (gfx1250) — matrix operand order repaired and checked by reading, not compiled, not run on hardware
//
#include <hip/hip_runtime.h>
#include <stdint.h>

typedef __attribute__((ext_vector_type(2)))  float    v2f;
typedef __attribute__((ext_vector_type(4)))  float    v4f;
typedef __attribute__((ext_vector_type(8)))  float    v8f;
typedef __attribute__((ext_vector_type(4)))  _Float16 v4h;
typedef __attribute__((ext_vector_type(16))) _Float16 v16h;
typedef __attribute__((ext_vector_type(4)))  unsigned int v4u;
typedef __attribute__((ext_vector_type(4)))  int      v4i;

#define EMBED 768
#define FFN   3072
#define NQ    4

// ---------------- f16 WMMA main kernel configuration ----------------
#define MBLK      64                  // rows per workgroup
#define NTHREADS  384                 // 12 waves: 2 M-blocks(32) x 6 N-blocks(128)
#define KSLAB     64                  // K staged per slab
#define NSLAB     (FFN / KSLAB)       // 48
#define QST       772                 // W2 quad-row col stride (8B entries); (8*QST)%64==32 -> conflict-free
#define HST       72                  // h quad-row col stride  (8B entries); (4*HST)%64==32 -> conflict-free
#define SLAB_QUADS (KSLAB / 4)        // 16
#define SLAB_BYTES (SLAB_QUADS * QST * 8)   // 98816
#define SLAB_CHUNKS (SLAB_BYTES / 16)       // 6176 x 16B
#define WS_QUADS  (FFN / 4)           // 768
#define WS_BYTES  ((size_t)WS_QUADS * QST * 8)  // 4,743,168

// LDS byte offsets (dynamic shared)
#define OFF_W2Q   0
#define W2Q_BYTES (2 * SLAB_BYTES)              // 197632 (double buffer)
#define OFF_HQ    (OFF_W2Q + W2Q_BYTES)
#define HQ_BYTES  (2 * SLAB_QUADS * HST * 8)    // 18432  (double buffer)
#define OFF_W1    (OFF_HQ + HQ_BYTES)
#define W1_BYTES  (NQ * FFN * 4)                // 49152
#define OFF_B1    (OFF_W1 + W1_BYTES)
#define B1_BYTES  (FFN * 4)                     // 12288
#define OFF_QS    (OFF_B1 + B1_BYTES)
#define QS_BYTES  (MBLK * NQ * 4)               // 1024
#define SMEM_MAIN (OFF_QS + QS_BYTES)           // 278528 < 320KB

// ---------------- CDNA5 async global->LDS (guarded) ----------------
#if __has_builtin(__builtin_amdgcn_global_load_async_to_lds_b128)
#define HAVE_ASYNC 1
#else
#define HAVE_ASYNC 0
#endif

typedef __attribute__((address_space(1))) v4i gas_v4i;   // global int4
typedef __attribute__((address_space(3))) v4i las_v4i;   // LDS int4

__device__ __forceinline__ void async_cp16(const char* g, char* l) {
#if HAVE_ASYNC
    __builtin_amdgcn_global_load_async_to_lds_b128(
        (gas_v4i*)g, (las_v4i*)l, 0, 0);
#endif
}

__device__ __forceinline__ void wait_async0() {
#if HAVE_ASYNC
#if __has_builtin(__builtin_amdgcn_s_wait_asynccnt)
    __builtin_amdgcn_s_wait_asynccnt(0);
#else
    asm volatile("s_wait_asynccnt 0x0" ::: "memory");
#endif
#endif
}

// copy one W2 slab (verbatim, already f16-quad-packed in ws) into an LDS buffer
__device__ __forceinline__ void copy_slab_issue(const char* __restrict__ ws, int slab,
                                                char* dstbuf, int tid) {
    const char* src = ws + (size_t)slab * SLAB_BYTES;
#if HAVE_ASYNC
    #pragma unroll
    for (int i = 0; i < 17; ++i) {
        int u = tid + i * NTHREADS;
        if (u < SLAB_CHUNKS) async_cp16(src + u * 16, dstbuf + u * 16);
    }
#else
    #pragma unroll
    for (int i = 0; i < 17; ++i) {
        int u = tid + i * NTHREADS;
        if (u < SLAB_CHUNKS) *(v4u*)(dstbuf + u * 16) = *(const v4u*)(src + u * 16);
    }
#endif
}

// cooperative h tile: 64 rows x KSLAB k -> f16, K-quad interleaved, into hq buffer
__device__ __forceinline__ void compute_h_slab(int slab, const float* w1sh, const float* b1sh,
                                               const float* qsh, char* hqbuf, int tid) {
    #pragma unroll
    for (int i = 0; i < 3; ++i) {
        int u = tid + i * NTHREADS;            // unit = (quad, row): 16*64 = 1024 units
        if (u < SLAB_QUADS * MBLK) {
            int qd = u >> 6;                   // local quad 0..15
            int m  = u & 63;                   // block row 0..63
            int kg = slab * KSLAB + qd * 4;    // global k of quad
            v4f qv = *(const v4f*)(qsh + m * NQ);
            v4f w0 = *(const v4f*)(w1sh + 0 * FFN + kg);
            v4f w1 = *(const v4f*)(w1sh + 1 * FFN + kg);
            v4f w2 = *(const v4f*)(w1sh + 2 * FFN + kg);
            v4f w3 = *(const v4f*)(w1sh + 3 * FFN + kg);
            v4f bb = *(const v4f*)(b1sh + kg);
            v4h hp;
            #pragma unroll
            for (int c = 0; c < 4; ++c) {
                float h = bb[c];
                h = fmaf(qv.x, w0[c], h);
                h = fmaf(qv.y, w1[c], h);
                h = fmaf(qv.z, w2[c], h);
                h = fmaf(qv.w, w3[c], h);
                h = fmaxf(h, 0.0f);
                hp[c] = (_Float16)h;
            }
            *(v4h*)(hqbuf + (size_t)(qd * HST + m) * 8) = hp;
        }
    }
}

// ---------------- repack: W2 f32 (3072x768) -> f16 K-quad-interleaved padded in ws ----------------
extern "C" __global__ __launch_bounds__(256)
void ffnq_repack_w2(const float* __restrict__ W2, char* __restrict__ ws) {
    int u = blockIdx.x * 256 + threadIdx.x;            // unit = (quad, col4)
    if (u >= WS_QUADS * (EMBED / 4)) return;
    int qd = u / (EMBED / 4);
    int c4 = u % (EMBED / 4);
    v4f r0 = *(const v4f*)(W2 + (size_t)(4 * qd + 0) * EMBED + 4 * c4);
    v4f r1 = *(const v4f*)(W2 + (size_t)(4 * qd + 1) * EMBED + 4 * c4);
    v4f r2 = *(const v4f*)(W2 + (size_t)(4 * qd + 2) * EMBED + 4 * c4);
    v4f r3 = *(const v4f*)(W2 + (size_t)(4 * qd + 3) * EMBED + 4 * c4);
    #pragma unroll
    for (int i = 0; i < 4; ++i) {
        v4h e;
        e[0] = (_Float16)r0[i]; e[1] = (_Float16)r1[i];
        e[2] = (_Float16)r2[i]; e[3] = (_Float16)r3[i];
        *(v4h*)(ws + ((size_t)qd * QST + 4 * c4 + i) * 8) = e;
    }
}

// ---------------- main fused kernel: f16 WMMA 16x16x32, f32 accumulate ----------------
extern "C" __global__ __launch_bounds__(NTHREADS)
void ffnq_fused_f16(const float* __restrict__ x,
                    const float* __restrict__ theta,
                    const float* __restrict__ W1,
                    const float* __restrict__ b1,
                    const char*  __restrict__ ws,   // repacked f16 W2
                    const float* __restrict__ b2,
                    float* __restrict__ out,
                    int M)
{
    extern __shared__ char smem[];
    char*  w2q  = smem + OFF_W2Q;
    char*  hq   = smem + OFF_HQ;
    float* w1sh = (float*)(smem + OFF_W1);
    float* b1sh = (float*)(smem + OFF_B1);
    float* qsh  = (float*)(smem + OFF_QS);

    const int tid     = threadIdx.x;
    const int lane    = tid & 31;
    const int wave    = tid >> 5;            // 0..11
    const int wm      = wave & 1;            // 32-row half
    const int wn      = wave >> 1;           // 0..5 : 128-col block
    const int halfsel = lane >> 4;
    const int mloc    = lane & 15;
    const int rowBase = blockIdx.x * MBLK;
    const int nBase   = wn * 128;

    // ---- Phase 0: stage W1, b1 (f32) and q = cos(x[:, :4])*cos(theta) into LDS ----
    #pragma unroll
    for (int i = 0; i < W1_BYTES / 16 / NTHREADS; ++i)   // 8
        ((v4f*)w1sh)[tid + i * NTHREADS] = ((const v4f*)W1)[tid + i * NTHREADS];
    #pragma unroll
    for (int i = 0; i < B1_BYTES / 16 / NTHREADS; ++i)   // 2
        ((v4f*)b1sh)[tid + i * NTHREADS] = ((const v4f*)b1)[tid + i * NTHREADS];
    if (tid < MBLK * NQ) {
        int r = tid >> 2, q = tid & 3;
        int row = rowBase + r; if (row >= M) row = M - 1;
        qsh[tid] = __builtin_cosf(x[(size_t)row * EMBED + q]) * __builtin_cosf(theta[q]);
    }
    __syncthreads();

    // ---- accumulators, bias-seeded (C layout: col depends only on lane%16) ----
    v8f acc[2][8];
    #pragma unroll
    for (int mt = 0; mt < 2; ++mt)
        #pragma unroll
        for (int t = 0; t < 8; ++t) {
            float bv = b2[nBase + t * 16 + mloc];
            acc[mt][t] = (v8f){bv, bv, bv, bv, bv, bv, bv, bv};
        }

    // ---- prologue: h(0) and W2 slab 0 ----
    compute_h_slab(0, w1sh, b1sh, qsh, hq, tid);
    copy_slab_issue(ws, 0, w2q, tid);
    wait_async0();
    __syncthreads();

    // ---- main pipeline over 48 slabs of K=64 ----
    for (int s = 0; s < NSLAB; ++s) {
        char* wqb = w2q + (s & 1) * SLAB_BYTES;
        char* hqb = hq  + (s & 1) * (SLAB_QUADS * HST * 8);
        if (s + 1 < NSLAB) {
            copy_slab_issue(ws, s + 1, w2q + ((s + 1) & 1) * SLAB_BYTES, tid);
            compute_h_slab(s + 1, w1sh, b1sh, qsh,
                           hq + ((s + 1) & 1) * (SLAB_QUADS * HST * 8), tid);
        }

        #pragma unroll
        for (int kk = 0; kk < KSLAB / 32; ++kk) {        // 2 k-steps of K=32
            const int qb = kk * 8;                       // local quad base
            // single base VGPR per k-step; all fragment loads use immediate DS offsets
            const char* abase = hqb + (size_t)((qb + 2 * halfsel) * HST + wm * 32 + mloc) * 8;
            const char* bbase = wqb + (size_t)((qb + 2 * halfsel) * QST + nBase + mloc) * 8;

            // A fragments: two 16x32 f16 tiles from hq (16-bit A layout, quad loads)
            v16h A[2];
            #pragma unroll
            for (int mt = 0; mt < 2; ++mt) {
                union { v16h v; v4h q[4]; } au;
                au.q[0] = *(v4h*)(abase + (size_t)(0 * HST + mt * 16) * 8);
                au.q[1] = *(v4h*)(abase + (size_t)(1 * HST + mt * 16) * 8);
                au.q[2] = *(v4h*)(abase + (size_t)(4 * HST + mt * 16) * 8);
                au.q[3] = *(v4h*)(abase + (size_t)(5 * HST + mt * 16) * 8);
                A[mt] = au.v;
            }
            // B fragments: 32x16 f16 per N tile; each reused by 2 WMMAs (M reuse)
            #pragma unroll
            for (int t = 0; t < 8; ++t) {
                union { v16h v; v4h q[4]; } bu;
                bu.q[0] = *(v4h*)(bbase + (size_t)(0 * QST + t * 16) * 8);
                bu.q[1] = *(v4h*)(bbase + (size_t)(1 * QST + t * 16) * 8);
                bu.q[2] = *(v4h*)(bbase + (size_t)(4 * QST + t * 16) * 8);
                bu.q[3] = *(v4h*)(bbase + (size_t)(5 * QST + t * 16) * 8);
                acc[0][t] = __builtin_amdgcn_wmma_f32_16x16x32_f16(
                    false, A[0], false, bu.v, (short)0, acc[0][t], false, false);
                acc[1][t] = __builtin_amdgcn_wmma_f32_16x16x32_f16(
                    false, A[1], false, bu.v, (short)0, acc[1][t], false, false);
            }
        }
        wait_async0();
        __syncthreads();
    }

    // ---- epilogue ----
    #pragma unroll
    for (int mt = 0; mt < 2; ++mt)
        #pragma unroll
        for (int t = 0; t < 8; ++t) {
            int col = nBase + t * 16 + mloc;
            #pragma unroll
            for (int r = 0; r < 8; ++r) {
                int row = rowBase + wm * 32 + mt * 16 + r + 8 * halfsel;
                if (row < M) out[(size_t)row * EMBED + col] = acc[mt][t][r];
            }
        }
}

// ---------------- fallback: round-1 f32 WMMA 16x16x4 kernel (used when ws too small) ----------------
#define F_MBLK 64
#define F_KSLAB 16
#define F_NT 24
#define F_STRIDE 776
#define F_NSLAB (FFN / F_KSLAB)
#define F_PREV4 ((F_KSLAB * EMBED / 4) / 256)

extern "C" __global__ __launch_bounds__(256)
void ffnq_fused_f32(const float* __restrict__ x, const float* __restrict__ theta,
                    const float* __restrict__ W1, const float* __restrict__ b1,
                    const float* __restrict__ W2, const float* __restrict__ b2,
                    float* __restrict__ out, int M)
{
    extern __shared__ float lds[];
    float* w2buf = lds;
    float* qsh   = lds + 2 * F_KSLAB * F_STRIDE;
    const int tid = threadIdx.x, lane = tid & 31, wave = tid >> 5;
    const int wm = wave & 3, wn = wave >> 2, halfsel = lane >> 4, mloc = lane & 15;
    const int rowBase = blockIdx.x * F_MBLK, waveRow = rowBase + wm * 16, nBase = wn * (F_NT * 16);
    {
        int r = tid >> 2, q = tid & 3;
        int row = rowBase + r; if (row >= M) row = M - 1;
        qsh[r * NQ + q] = __builtin_cosf(x[(size_t)row * EMBED + q]) * __builtin_cosf(theta[q]);
    }
    {
        const v4f* s4 = (const v4f*)W2;
        #pragma unroll
        for (int j = 0; j < F_PREV4; ++j) {
            int i4 = tid + j * 256, r = i4 / (EMBED / 4), c4 = i4 % (EMBED / 4);
            *(v4f*)(w2buf + r * F_STRIDE + c4 * 4) = s4[i4];
        }
    }
    v8f acc[F_NT];
    #pragma unroll
    for (int t = 0; t < F_NT; ++t) {
        float bv = b2[nBase + t * 16 + mloc];
        acc[t] = (v8f){bv, bv, bv, bv, bv, bv, bv, bv};
    }
    __syncthreads();
    const float q0 = qsh[(wm * 16 + mloc) * NQ + 0], q1 = qsh[(wm * 16 + mloc) * NQ + 1];
    const float q2 = qsh[(wm * 16 + mloc) * NQ + 2], q3 = qsh[(wm * 16 + mloc) * NQ + 3];
    for (int s = 0; s < F_NSLAB; ++s) {
        const float* wb = w2buf + (s & 1) * (F_KSLAB * F_STRIDE);
        v4f pre[F_PREV4];
        const bool havenext = (s + 1 < F_NSLAB);
        if (havenext) {
            const v4f* s4 = (const v4f*)(W2 + (size_t)(s + 1) * F_KSLAB * EMBED);
            #pragma unroll
            for (int j = 0; j < F_PREV4; ++j) pre[j] = s4[tid + j * 256];
        }
        #pragma unroll
        for (int kk = 0; kk < F_KSLAB / 4; ++kk) {
            const int kloc = kk * 4 + 2 * halfsel, kg = s * F_KSLAB + kloc;
            float h0 = b1[kg];
            h0 = fmaf(q0, W1[0 * FFN + kg], h0); h0 = fmaf(q1, W1[1 * FFN + kg], h0);
            h0 = fmaf(q2, W1[2 * FFN + kg], h0); h0 = fmaf(q3, W1[3 * FFN + kg], h0);
            h0 = fmaxf(h0, 0.0f);
            float h1 = b1[kg + 1];
            h1 = fmaf(q0, W1[0 * FFN + kg + 1], h1); h1 = fmaf(q1, W1[1 * FFN + kg + 1], h1);
            h1 = fmaf(q2, W1[2 * FFN + kg + 1], h1); h1 = fmaf(q3, W1[3 * FFN + kg + 1], h1);
            h1 = fmaxf(h1, 0.0f);
            v2f a; a.x = h0; a.y = h1;
            #pragma unroll
            for (int t = 0; t < F_NT; ++t) {
                const int col = nBase + t * 16 + mloc;
                v2f b;
                b.x = wb[(kloc + 0) * F_STRIDE + col];
                b.y = wb[(kloc + 1) * F_STRIDE + col];
                acc[t] = __builtin_amdgcn_wmma_f32_16x16x4_f32(false, a, false, b, (short)0, acc[t], false, false);
            }
        }
        if (havenext) {
            float* dst = w2buf + ((s + 1) & 1) * (F_KSLAB * F_STRIDE);
            #pragma unroll
            for (int j = 0; j < F_PREV4; ++j) {
                int i4 = tid + j * 256, r = i4 / (EMBED / 4), c4 = i4 % (EMBED / 4);
                *(v4f*)(dst + r * F_STRIDE + c4 * 4) = pre[j];
            }
        }
        __syncthreads();
    }
    #pragma unroll
    for (int t = 0; t < F_NT; ++t) {
        const int col = nBase + t * 16 + mloc;
        #pragma unroll
        for (int r = 0; r < 8; ++r) {
            const int row = waveRow + r + 8 * halfsel;
            if (row < M) out[(size_t)row * EMBED + col] = acc[t][r];
        }
    }
}

extern "C" void kernel_launch(void* const* d_in, const int* in_sizes, int n_in,
                              void* d_out, int out_size, void* d_ws, size_t ws_size,
                              hipStream_t stream) {
    const float* x     = (const float*)d_in[0];
    const float* theta = (const float*)d_in[1];
    const float* W1    = (const float*)d_in[2];
    const float* b1    = (const float*)d_in[3];
    const float* W2    = (const float*)d_in[4];
    const float* b2    = (const float*)d_in[5];
    float* out = (float*)d_out;
    const int M = in_sizes[0] / EMBED;          // 16384
    const int grid = (M + MBLK - 1) / MBLK;     // 256

    if (ws_size >= WS_BYTES) {
        // repack W2 once per call (deterministic), then fused f16-WMMA kernel
        const int units = WS_QUADS * (EMBED / 4);
        ffnq_repack_w2<<<(units + 255) / 256, 256, 0, stream>>>(W2, (char*)d_ws);
        ffnq_fused_f16<<<grid, NTHREADS, SMEM_MAIN, stream>>>(
            x, theta, W1, b1, (const char*)d_ws, b2, out, M);
    } else {
        const size_t shmem = (2 * F_KSLAB * F_STRIDE + F_MBLK * NQ) * sizeof(float);
        ffnq_fused_f32<<<grid, 256, shmem, stream>>>(x, theta, W1, b1, W2, b2, out, M);
    }
}
